// CrystalGraphEncoder_10496900071649
// MI455X (gfx1250) — matrix operand order repaired and checked by reading, not compiled, not run on hardware
//
#include <hip/hip_runtime.h>
#include <hip/hip_bf16.h>
#include <math.h>

// Problem sizes (fixed by the reference)
#define NN       20000
#define NE       320000
#define NG       64
#define H        256
#define NODE_DIM 92
#define NODE_PAD 96
#define EDGE_DIM 64
#define LATENT   128
#define NLAYERS  6

typedef __attribute__((ext_vector_type(16))) __bf16 v16bf;
typedef __attribute__((ext_vector_type(8)))  __bf16 v8bf;
typedef __attribute__((ext_vector_type(8)))  float  v8f;

enum { MODE_PLAIN = 0, MODE_MSG = 1, MODE_UPD = 2 };
enum { EPI_STORE_BF16 = 0, EPI_SCATTER = 1, EPI_LN = 2, EPI_STORE_F32 = 3 };

struct FusedArgs {
  const __bf16* A; int lda;              // MODE_PLAIN source
  const __bf16* h; const __bf16* e;      // MODE_MSG / MODE_UPD sources
  const float*  agg;                     // MODE_UPD second half (f32)
  const int* srcIdx; const int* dstIdx;  // gather indices (MODE_MSG)
  const __bf16* W1p; const float* b1;    // packed layer-1 weights (K1 x 256)
  const __bf16* W2p; const float* b2;    // packed layer-2 weights (256 x 16*NT2)
  float* outF; __bf16* outB; int ldo;    // store epilogues
  float* aggOut; const int* tgt;         // scatter epilogue
  __bf16* hio; const float* gamma; const float* beta; // LN epilogue (in/out h)
  int M; int K1;
};

// ---- CDNA5 async global->LDS copy (no VGPR staging, tracked by ASYNCcnt) ----
__device__ __forceinline__ void async_ld16(unsigned lds_off, unsigned long long gaddr) {
  asm volatile("global_load_async_to_lds_b128 %0, %1, off"
               :: "v"(lds_off), "v"(gaddr) : "memory");
}
__device__ __forceinline__ void wait_async0() {
  asm volatile("s_wait_asynccnt 0x0" ::: "memory");
}

__device__ __forceinline__ v16bf combine8(v8bf lo, v8bf hi) {
  v16bf r;
#pragma unroll
  for (int i = 0; i < 8; ++i) { r[i] = lo[i]; r[i + 8] = hi[i]; }
  return r;
}

// A-fragment loader. ISA 16-bit A layout (16x32):
//   lanes 0-15 : row = lane,    elems 0-7 => K = 0..7,  elems 8-15 => K = 16..23
//   lanes 16-31: row = lane-16, elems 0-7 => K = 8..15, elems 8-15 => K = 24..31
// half8 = (lane&16)?8:0 ; runs are 8 contiguous elements => two 16B loads.
template <int MODE>
__device__ __forceinline__ v16bf load_afrag(const FusedArgs& p, int row, int s, int half8) {
  if (MODE == MODE_PLAIN) {
    const __bf16* q = p.A + (size_t)row * p.lda + s * 32 + half8;
    return combine8(*(const v8bf*)q, *(const v8bf*)(q + 16));
  } else if (MODE == MODE_MSG) {
    // fused gather-concat: K [0,256)=h[src], [256,512)=h[dst], [512,768)=e[row]
    const __bf16* base;
    if (s < 8)       base = p.h + (size_t)p.srcIdx[row] * H + s * 32;
    else if (s < 16) base = p.h + (size_t)p.dstIdx[row] * H + (s - 8) * 32;
    else             base = p.e + (size_t)row * H + (s - 16) * 32;
    const __bf16* q = base + half8;
    return combine8(*(const v8bf*)q, *(const v8bf*)(q + 16));
  } else { // MODE_UPD: K [0,256)=h[row] (bf16), [256,512)=agg[row] (f32 -> bf16)
    if (s < 8) {
      const __bf16* q = p.h + (size_t)row * H + s * 32 + half8;
      return combine8(*(const v8bf*)q, *(const v8bf*)(q + 16));
    } else {
      const float* q = p.agg + (size_t)row * H + (s - 8) * 32 + half8;
      v8f f0 = *(const v8f*)q;
      v8f f1 = *(const v8f*)(q + 16);
      v16bf r;
#pragma unroll
      for (int i = 0; i < 8; ++i) { r[i] = (__bf16)f0[i]; r[i + 8] = (__bf16)f1[i]; }
      return r;
    }
  }
}

// Fully fused 2-layer MLP:  OUT = (silu(A @ W1 + b1)) @ W2 + b2
// Block = 128 rows x 256 hidden, 8 waves, each wave owns a 16-row strip and
// all 16 N-tiles (v_wmma_f32_16x16x32_bf16, f32 accumulate).
// Weight K-step tiles ping-pong through 2 x 16KB LDS buffers, filled with
// GLOBAL_LOAD_ASYNC_TO_LDS_B128 issued before the WMMA chain of the current
// step — async DMA hides the global latency with zero VGPR staging cost.
template <int MODE, int EPI, int NT2>
__global__ __launch_bounds__(256, 1)
void fused_mlp(FusedArgs p) {
  __shared__ __attribute__((aligned(32))) __bf16 lds_b[2][16 * 32 * 16]; // 2 x 16 KB ping-pong
  __shared__ __attribute__((aligned(32))) __bf16 lds_h[128 * H];         // 64 KB hidden acts

  const int lane   = threadIdx.x & 31;
  const int wave   = threadIdx.x >> 5;
  const int half8  = (lane & 16) ? 8 : 0;
  const int mrow   = lane & 15;
  const int rowTop = blockIdx.x * 128;
  const int rowg   = rowTop + wave * 16 + mrow;
  const int rowc   = (rowg < p.M) ? rowg : (p.M - 1);   // clamp for loads, mask at stores

  // LDS byte offsets of the two ping-pong buffers (low 32 bits of the
  // generic pointer are the LDS address — LDS aperture truncation)
  const unsigned lb0 = (unsigned)(size_t)(&lds_b[0][0]);
  const unsigned lb1 = (unsigned)(size_t)(&lds_b[1][0]);

  v8f acc[16];
  const v8f zero = {0.f, 0.f, 0.f, 0.f, 0.f, 0.f, 0.f, 0.f};
#pragma unroll
  for (int t = 0; t < 16; ++t) acc[t] = zero;

  // ---------------- phase 1: HID = silu(A @ W1 + b1), N = 256 ----------------
  const int ns1 = p.K1 >> 5;
  {
    const unsigned long long g = (unsigned long long)(size_t)p.W1p;
#pragma unroll
    for (int j = 0; j < 4; ++j) {
      const unsigned idx = threadIdx.x + j * 256;
      async_ld16(lb0 + idx * 16u, g + (unsigned long long)idx * 16ull);
    }
  }
  wait_async0();
  __syncthreads();
  for (int s = 0; s < ns1; ++s) {
    const int buf = s & 1;
    // kick off async DMA of the next weight tile into the other buffer
    if (s + 1 < ns1) {
      const unsigned long long g = (unsigned long long)(size_t)(p.W1p + (size_t)(s + 1) * 8192);
      const unsigned lb = buf ? lb0 : lb1;
#pragma unroll
      for (int j = 0; j < 4; ++j) {
        const unsigned idx = threadIdx.x + j * 256;
        async_ld16(lb + idx * 16u, g + (unsigned long long)idx * 16ull);
      }
    }
    v16bf af = load_afrag<MODE>(p, rowc, s, half8);
#pragma unroll
    for (int t = 0; t < 16; ++t) {
      v16bf bf = *(const v16bf*)(lds_b[buf] + (t * 32 + lane) * 16); // contiguous 32B frag
      acc[t] = __builtin_amdgcn_wmma_f32_16x16x32_bf16(false, af, false, bf,
                                                       (short)0, acc[t], false, false);
    }
    wait_async0();
    __syncthreads();
  }

  // bias + SiLU -> LDS (bf16 row-major [128][256]); C layout: row = r + half8
#pragma unroll
  for (int t = 0; t < 16; ++t) {
    const int col = mrow + t * 16;
    const float bv = p.b1[col];
#pragma unroll
    for (int r = 0; r < 8; ++r) {
      const int rloc = wave * 16 + r + half8;
      float v = acc[t][r] + bv;
      v = v / (1.f + __expf(-v));                 // SiLU
      lds_h[rloc * H + col] = (__bf16)v;
    }
  }

  // ---------------- phase 2: OUT = HID @ W2 + b2, K = 256 ----------------
#pragma unroll
  for (int t = 0; t < 16; ++t) acc[t] = zero;
  constexpr int J2 = NT2 / 4;   // b128 per thread per K-step (NT2*64 / 256 threads)
  {
    const unsigned long long g = (unsigned long long)(size_t)p.W2p;
#pragma unroll
    for (int j = 0; j < J2; ++j) {
      const unsigned idx = threadIdx.x + j * 256;
      async_ld16(lb0 + idx * 16u, g + (unsigned long long)idx * 16ull);
    }
  }
  wait_async0();
  __syncthreads();   // also covers lds_h writes above
  for (int s = 0; s < 8; ++s) {
    const int buf = s & 1;
    if (s + 1 < 8) {
      const unsigned long long g =
          (unsigned long long)(size_t)(p.W2p + (size_t)(s + 1) * (NT2 * 512));
      const unsigned lb = buf ? lb0 : lb1;
#pragma unroll
      for (int j = 0; j < J2; ++j) {
        const unsigned idx = threadIdx.x + j * 256;
        async_ld16(lb + idx * 16u, g + (unsigned long long)idx * 16ull);
      }
    }
    const __bf16* q = lds_h + (wave * 16 + mrow) * H + s * 32 + half8;
    v16bf af = combine8(*(const v8bf*)q, *(const v8bf*)(q + 16));
#pragma unroll
    for (int t = 0; t < NT2; ++t) {
      v16bf bf = *(const v16bf*)(lds_b[buf] + (t * 32 + lane) * 16);
      acc[t] = __builtin_amdgcn_wmma_f32_16x16x32_bf16(false, af, false, bf,
                                                       (short)0, acc[t], false, false);
    }
    wait_async0();
    __syncthreads();
  }

  // ---------------- epilogue ----------------
  if (EPI == EPI_STORE_BF16) {
#pragma unroll
    for (int t = 0; t < NT2; ++t) {
      const int col = mrow + t * 16;
      const float bv = p.b2[col];
#pragma unroll
      for (int r = 0; r < 8; ++r) {
        const int row = rowTop + wave * 16 + r + half8;
        if (row < p.M) p.outB[(size_t)row * p.ldo + col] = (__bf16)(acc[t][r] + bv);
      }
    }
  } else if (EPI == EPI_STORE_F32) {
#pragma unroll
    for (int t = 0; t < NT2; ++t) {
      const int col = mrow + t * 16;
      const float bv = p.b2[col];
#pragma unroll
      for (int r = 0; r < 8; ++r) {
        const int row = rowTop + wave * 16 + r + half8;
        if (row < p.M) p.outF[(size_t)row * p.ldo + col] = acc[t][r] + bv;
      }
    }
  } else if (EPI == EPI_SCATTER) {
    // segment_sum(messages, target): scatter-add straight from registers
#pragma unroll
    for (int r = 0; r < 8; ++r) {
      const int row = rowTop + wave * 16 + r + half8;
      if (row < p.M) {
        const size_t base = (size_t)p.tgt[row] * H;
#pragma unroll
        for (int t = 0; t < NT2; ++t) {
          const int col = mrow + t * 16;
          atomicAdd(p.aggOut + base + col, acc[t][r] + p.b2[col]);
        }
      }
    }
  } else { // EPI_LN: h = LayerNorm(h + update_out)
#pragma unroll
    for (int t = 0; t < 16; ++t) {
      const int col = mrow + t * 16;
      const float bv = p.b2[col];
#pragma unroll
      for (int r = 0; r < 8; ++r) {
        const int row = rowTop + wave * 16 + r + half8;
        const float res = (row < p.M) ? (float)p.hio[(size_t)row * H + col] : 0.f;
        acc[t][r] = acc[t][r] + bv + res;
      }
    }
    // per-row stats: 16 local values + shfl_xor over the 16 lanes of this half
#pragma unroll
    for (int r = 0; r < 8; ++r) {
      float sm = 0.f, sq = 0.f;
#pragma unroll
      for (int t = 0; t < 16; ++t) { const float v = acc[t][r]; sm += v; sq += v * v; }
#pragma unroll
      for (int m = 1; m < 16; m <<= 1) {
        sm += __shfl_xor(sm, m, 32);
        sq += __shfl_xor(sq, m, 32);
      }
      const float mu  = sm * (1.f / 256.f);
      const float var = sq * (1.f / 256.f) - mu * mu;
      const float inv = rsqrtf(var + 1e-5f);
      const int row = rowTop + wave * 16 + r + half8;
      if (row < p.M) {
#pragma unroll
        for (int t = 0; t < 16; ++t) {
          const int col = mrow + t * 16;
          const float vo = (acc[t][r] - mu) * inv * p.gamma[col] + p.beta[col];
          p.hio[(size_t)row * H + col] = (__bf16)vo;
        }
      }
    }
  }
}

// Pre-pack W [K,N] f32 row-major -> bf16 in exact B-fragment order:
// packed[((s*(N/16)+t)*32 + lane)*16 + e] = W[k][n],
//   k = 32s + 8*(lane>=16) + (e&7) + 16*(e>=8),  n = 16t + (lane&15).   (ISA 16-bit B layout)
__global__ void pack_w(const float* __restrict__ W, __bf16* __restrict__ P,
                       int K, int N, int Kpad) {
  const int idx = blockIdx.x * 256 + threadIdx.x;
  const int total = Kpad * N;
  if (idx >= total) return;
  const int e  = idx & 15;
  const int l  = (idx >> 4) & 31;
  const int nt = N >> 4;
  const int t  = (idx >> 9) % nt;
  const int s  = idx / (nt << 9);
  const int k  = (s << 5) + ((l & 16) ? 8 : 0) + (e & 7) + ((e & 8) ? 16 : 0);
  const int n  = (t << 4) + (l & 15);
  P[idx] = (__bf16)((k < K) ? W[(size_t)k * N + n] : 0.f);
}

__global__ void cvt_pad(const float* __restrict__ X, __bf16* __restrict__ Y,
                        int M, int Kin, int Kout) {
  const long long idx = (long long)blockIdx.x * 256 + threadIdx.x;
  if (idx >= (long long)M * Kout) return;
  const int k = (int)(idx % Kout);
  const long long m = idx / Kout;
  Y[idx] = (k < Kin) ? (__bf16)X[m * Kin + k] : (__bf16)0.f;
}

__device__ __forceinline__ void atomicMaxF(float* addr, float v) {
  if (v >= 0.f) atomicMax((int*)addr, __float_as_int(v));
  else          atomicMin((unsigned int*)addr, (unsigned int)__float_as_int(v));
}

__global__ void pool_init(float* sums, float* maxv, float* cnts) {
  const int i = blockIdx.x * 256 + threadIdx.x;
  if (i < NG * H) { sums[i] = 0.f; maxv[i] = -3.402823466e38f; }
  if (i < NG) cnts[i] = 0.f;
}

__global__ void pool_accum(const __bf16* __restrict__ h, const int* __restrict__ batch,
                           float* sums, float* maxv, float* cnts, int M) {
  const long long i = (long long)blockIdx.x * 256 + threadIdx.x;
  if (i >= (long long)M * H) return;
  const int c = (int)(i & (H - 1));
  const int n = (int)(i >> 8);
  const int g = batch[n];
  const float v = (float)h[i];
  atomicAdd(sums + (size_t)g * H + c, v);
  atomicMaxF(maxv + (size_t)g * H + c, v);
  if (c == 0) atomicAdd(cnts + g, 1.f);
}

__global__ void pool_finalize(const float* sums, const float* maxv, const float* cnts,
                              float* __restrict__ grF, __bf16* __restrict__ grB) {
  const int i = blockIdx.x * 256 + threadIdx.x;
  if (i >= NG * 2 * H) return;
  const int c = i & (2 * H - 1);
  const int g = i >> 9;
  const float cnt = cnts[g];
  float v;
  if (c < H) v = sums[(size_t)g * H + c] / fmaxf(cnt, 1.f);
  else       v = (cnt > 0.f) ? maxv[(size_t)g * H + (c - H)] : 0.f;
  grF[i] = v;
  grB[i] = (__bf16)v;
}

extern "C" void kernel_launch(void* const* d_in, const int* in_sizes, int n_in,
                              void* d_out, int out_size, void* d_ws, size_t ws_size,
                              hipStream_t stream) {
  (void)in_sizes; (void)n_in; (void)out_size; (void)ws_size;

  const float* x     = (const float*)d_in[0];
  const int*   eidx  = (const int*)  d_in[1];
  const float* eattr = (const float*)d_in[2];
  const int*   batch = (const int*)  d_in[3];

  // params flattened in JAX pytree order (dict keys sorted alphabetically):
  // edge_proj{l1{b,w},l2{b,w}}, layers[0..5]{message{l1{b,w},l2{b,w}},
  //   norm{b,g}, update{l1{b,w},l2{b,w}}}, node_proj{...}, out_proj{...}
  int pi = 4;
  auto F = [&]() { return (const float*)d_in[pi++]; };
  const float *ep1b = F(), *ep1w = F(), *ep2b = F(), *ep2w = F();
  const float *m1b[6], *m1w[6], *m2b[6], *m2w[6], *nb[6], *ng[6],
              *u1b[6], *u1w[6], *u2b[6], *u2w[6];
  for (int i = 0; i < 6; ++i) {
    m1b[i] = F(); m1w[i] = F(); m2b[i] = F(); m2w[i] = F();
    nb[i]  = F(); ng[i]  = F();
    u1b[i] = F(); u1w[i] = F(); u2b[i] = F(); u2w[i] = F();
  }
  const float *np1b = F(), *np1w = F(), *np2b = F(), *np2w = F();
  const float *op1b = F(), *op1w = F(), *op2b = F(), *op2w = F();

  const int* src = eidx;
  const int* dst = eidx + NE;

  // workspace bump allocator (256B aligned)
  char* wp = (char*)d_ws;
  auto alloc = [&](size_t bytes) -> void* {
    void* r = (void*)wp;
    wp += (bytes + 255) & ~(size_t)255;
    return r;
  };
  __bf16* hbuf = (__bf16*)alloc((size_t)NN * H * 2);
  __bf16* ebuf = (__bf16*)alloc((size_t)NE * H * 2);
  __bf16* xpad = (__bf16*)alloc((size_t)NN * NODE_PAD * 2);
  __bf16* eabf = (__bf16*)alloc((size_t)NE * EDGE_DIM * 2);
  float*  agg  = (float*) alloc((size_t)NN * H * 4);
  float*  sums = (float*) alloc((size_t)NG * H * 4);
  float*  maxv = (float*) alloc((size_t)NG * H * 4);
  float*  cnts = (float*) alloc((size_t)NG * 4);
  __bf16* grb  = (__bf16*)alloc((size_t)NG * 2 * H * 2);
  __bf16* pw_n1 = (__bf16*)alloc((size_t)NODE_PAD * H * 2);
  __bf16* pw_n2 = (__bf16*)alloc((size_t)H * H * 2);
  __bf16* pw_e1 = (__bf16*)alloc((size_t)EDGE_DIM * H * 2);
  __bf16* pw_e2 = (__bf16*)alloc((size_t)H * H * 2);
  __bf16 *pw_m1[6], *pw_m2[6], *pw_u1[6], *pw_u2[6];
  for (int i = 0; i < 6; ++i) {
    pw_m1[i] = (__bf16*)alloc((size_t)3 * H * H * 2);
    pw_m2[i] = (__bf16*)alloc((size_t)H * H * 2);
    pw_u1[i] = (__bf16*)alloc((size_t)2 * H * H * 2);
    pw_u2[i] = (__bf16*)alloc((size_t)H * H * 2);
  }
  __bf16* pw_o1 = (__bf16*)alloc((size_t)2 * H * H * 2);
  __bf16* pw_o2 = (__bf16*)alloc((size_t)H * LATENT * 2);

  // ---- one-time weight packing & input conversion ----
  auto pack = [&](const float* W, __bf16* P, int K, int N, int Kpad) {
    const int total = Kpad * N;
    pack_w<<<(total + 255) / 256, 256, 0, stream>>>(W, P, K, N, Kpad);
  };
  pack(np1w, pw_n1, NODE_DIM, H, NODE_PAD);
  pack(np2w, pw_n2, H, H, H);
  pack(ep1w, pw_e1, EDGE_DIM, H, EDGE_DIM);
  pack(ep2w, pw_e2, H, H, H);
  for (int i = 0; i < 6; ++i) {
    pack(m1w[i], pw_m1[i], 3 * H, H, 3 * H);
    pack(m2w[i], pw_m2[i], H, H, H);
    pack(u1w[i], pw_u1[i], 2 * H, H, 2 * H);
    pack(u2w[i], pw_u2[i], H, H, H);
  }
  pack(op1w, pw_o1, 2 * H, H, 2 * H);
  pack(op2w, pw_o2, H, LATENT, H);

  cvt_pad<<<(NN * NODE_PAD + 255) / 256, 256, 0, stream>>>(x, xpad, NN, NODE_DIM, NODE_PAD);
  cvt_pad<<<(NE * EDGE_DIM + 255) / 256, 256, 0, stream>>>(eattr, eabf, NE, EDGE_DIM, EDGE_DIM);

  // ---- node / edge projections ----
  {
    FusedArgs fa = {};
    fa.A = xpad; fa.lda = NODE_PAD;
    fa.W1p = pw_n1; fa.b1 = np1b; fa.W2p = pw_n2; fa.b2 = np2b;
    fa.outB = hbuf; fa.ldo = H; fa.M = NN; fa.K1 = NODE_PAD;
    fused_mlp<MODE_PLAIN, EPI_STORE_BF16, 16><<<(NN + 127) / 128, 256, 0, stream>>>(fa);
  }
  {
    FusedArgs fa = {};
    fa.A = eabf; fa.lda = EDGE_DIM;
    fa.W1p = pw_e1; fa.b1 = ep1b; fa.W2p = pw_e2; fa.b2 = ep2b;
    fa.outB = ebuf; fa.ldo = H; fa.M = NE; fa.K1 = EDGE_DIM;
    fused_mlp<MODE_PLAIN, EPI_STORE_BF16, 16><<<NE / 128, 256, 0, stream>>>(fa);
  }

  // ---- 6 message-passing layers ----
  for (int i = 0; i < NLAYERS; ++i) {
    hipMemsetAsync(agg, 0, (size_t)NN * H * 4, stream);
    {
      FusedArgs fa = {};
      fa.h = hbuf; fa.e = ebuf; fa.srcIdx = src; fa.dstIdx = dst;
      fa.W1p = pw_m1[i]; fa.b1 = m1b[i]; fa.W2p = pw_m2[i]; fa.b2 = m2b[i];
      fa.aggOut = agg; fa.tgt = dst;
      fa.M = NE; fa.K1 = 3 * H;
      fused_mlp<MODE_MSG, EPI_SCATTER, 16><<<NE / 128, 256, 0, stream>>>(fa);
    }
    {
      FusedArgs fa = {};
      fa.h = hbuf; fa.agg = agg;
      fa.W1p = pw_u1[i]; fa.b1 = u1b[i]; fa.W2p = pw_u2[i]; fa.b2 = u2b[i];
      fa.hio = hbuf; fa.gamma = ng[i]; fa.beta = nb[i];
      fa.M = NN; fa.K1 = 2 * H;
      fused_mlp<MODE_UPD, EPI_LN, 16><<<(NN + 127) / 128, 256, 0, stream>>>(fa);
    }
  }

  // ---- pooling + out_proj ----
  pool_init<<<(NG * H + 255) / 256, 256, 0, stream>>>(sums, maxv, cnts);
  pool_accum<<<(NN * H + 255) / 256, 256, 0, stream>>>(hbuf, batch, sums, maxv, cnts, NN);

  float* out_latent = (float*)d_out;            // [64,128]
  float* out_gr     = out_latent + NG * LATENT; // [64,512]
  pool_finalize<<<(NG * 2 * H + 255) / 256, 256, 0, stream>>>(sums, maxv, cnts, out_gr, grb);

  {
    FusedArgs fa = {};
    fa.A = grb; fa.lda = 2 * H;
    fa.W1p = pw_o1; fa.b1 = op1b; fa.W2p = pw_o2; fa.b2 = op2b;
    fa.outF = out_latent; fa.ldo = LATENT;
    fa.M = NG; fa.K1 = 2 * H;
    fused_mlp<MODE_PLAIN, EPI_STORE_F32, 8><<<1, 256, 0, stream>>>(fa);
  }
}
